// BasicTransformerBlock_66254165508572
// MI455X (gfx1250) — compile-verified, matrix operand order checked by reading, not run on hardware
//
#include <hip/hip_runtime.h>
#include <hip/hip_bf16.h>

// ---------------------------------------------------------------------------
// BasicTransformerBlock for MI455X (gfx1250), bf16 WMMA everywhere.
//   x:[2,2048,1024] f32, ctx:[2,77,768] f32 -> out f32 [2,2048,1024]
// Compute-bound (~190 GFLOP over ~100MB) => route all GEMMs + attention
// through v_wmma_f32_16x16x32_bf16, fp32 accumulation.
// Workspace requirement: ~104 MiB.
// ---------------------------------------------------------------------------

typedef __bf16 bf16_t;
typedef bf16_t v16bf __attribute__((ext_vector_type(16)));
typedef bf16_t v8bf  __attribute__((ext_vector_type(8)));
typedef float  v8f   __attribute__((ext_vector_type(8)));

// Load one 16-element bf16 WMMA fragment (A- or B-style) from a row-major
// source. For 16-bit fragments, lane (lo/hi half) maps its 16 values to two
// contiguous 8-element runs: k = hi*8 + [0..7]  and  k = 16 + hi*8 + [0..7].
__device__ __forceinline__ v16bf load_frag(const bf16_t* __restrict__ p, int hi) {
  const v8bf lo = *(const v8bf*)(p + hi * 8);
  const v8bf hh = *(const v8bf*)(p + hi * 8 + 16);
  return __builtin_shufflevector(lo, hh, 0, 1, 2, 3, 4, 5, 6, 7,
                                 8, 9, 10, 11, 12, 13, 14, 15);
}

__device__ __forceinline__ v8f wmma_bf16(v16bf a, v16bf b, v8f c) {
  return __builtin_amdgcn_wmma_f32_16x16x32_bf16(
      /*neg_a=*/false, a, /*neg_b=*/false, b,
      /*c_mod=*/(short)0, c, /*reuse_a=*/false, /*reuse_b=*/false);
}

// ---------------------------------------------------------------------------
// Weight convert+transpose:  W[K][N] f32  ->  Wt[N][K] bf16
// ---------------------------------------------------------------------------
__global__ void __launch_bounds__(256) k_wt(const float* __restrict__ W,
                                            bf16_t* __restrict__ Wt,
                                            int K, int N) {
  int idx = blockIdx.x * 256 + threadIdx.x;
  if (idx >= K * N) return;
  int n = idx / K, k = idx - n * K;
  Wt[idx] = (bf16_t)W[(size_t)k * N + n];
}

// Elementwise f32 -> bf16
__global__ void __launch_bounds__(256) k_cvt(const float* __restrict__ in,
                                             bf16_t* __restrict__ out, int n) {
  int idx = blockIdx.x * 256 + threadIdx.x;
  if (idx < n) out[idx] = (bf16_t)in[idx];
}

// ---------------------------------------------------------------------------
// LayerNorm (dim=1024), one row per 256-thread block, bf16 output.
// ---------------------------------------------------------------------------
__global__ void __launch_bounds__(256) k_layernorm(const float* __restrict__ x,
                                                   const float* __restrict__ g,
                                                   const float* __restrict__ b,
                                                   bf16_t* __restrict__ y) {
  const int dim = 1024;
  __shared__ float sS[8], sQ[8];
  const int row = blockIdx.x;
  const float* xr = x + (size_t)row * dim;
  float s = 0.f, q = 0.f;
  for (int i = threadIdx.x; i < dim; i += 256) {
    float v = xr[i];
    s += v;
    q += v * v;
  }
  for (int m = 1; m < 32; m <<= 1) {
    s += __shfl_xor(s, m, 32);
    q += __shfl_xor(q, m, 32);
  }
  const int lane = threadIdx.x & 31, wid = threadIdx.x >> 5;
  if (lane == 0) { sS[wid] = s; sQ[wid] = q; }
  __syncthreads();
  if (wid == 0) {
    s = (lane < 8) ? sS[lane] : 0.f;
    q = (lane < 8) ? sQ[lane] : 0.f;
    for (int m = 1; m < 8; m <<= 1) {
      s += __shfl_xor(s, m, 32);
      q += __shfl_xor(q, m, 32);
    }
    if (lane == 0) { sS[0] = s; sQ[0] = q; }
  }
  __syncthreads();
  const float mean = sS[0] * (1.f / dim);
  const float var = sQ[0] * (1.f / dim) - mean * mean;
  const float rstd = rsqrtf(var + 1e-5f);
  for (int i = threadIdx.x; i < dim; i += 256) {
    float v = (xr[i] - mean) * rstd * g[i] + b[i];
    y[(size_t)row * dim + i] = (bf16_t)v;
  }
}

// ---------------------------------------------------------------------------
// V transpose:  v[b*mRows][1024] bf16 -> vt[b][16][64][mpad] bf16 (zero pad)
// ---------------------------------------------------------------------------
__global__ void __launch_bounds__(256) k_vt(const bf16_t* __restrict__ v,
                                            bf16_t* __restrict__ vt,
                                            int mRows, int mpad) {
  int idx = blockIdx.x * 256 + threadIdx.x;
  int total = 2 * 16 * 64 * mpad;
  if (idx >= total) return;
  int key = idx % mpad;
  int t = idx / mpad;
  int d = t & 63; t >>= 6;
  int hh = t & 15; int bb = t >> 4;
  bf16_t val = (bf16_t)0.f;
  if (key < mRows) val = v[(size_t)(bb * mRows + key) * 1024 + hh * 64 + d];
  vt[idx] = val;
}

// ---------------------------------------------------------------------------
// GEMM  C = A[M,K](bf16) x Bt[N,K](bf16), 8 waves/block (4M x 2N),
// each wave: 16(M) x 64(N) via 4 WMMAs per 32-deep k-step.
// ---------------------------------------------------------------------------
__global__ void __launch_bounds__(256) k_gemm_bf16(const bf16_t* __restrict__ A,
                                                   const bf16_t* __restrict__ Bt,
                                                   bf16_t* __restrict__ C,
                                                   int M, int N, int K,
                                                   float scale) {
  const int lane = threadIdx.x & 31, wid = threadIdx.x >> 5;
  const int hi = (lane >> 4) & 1, l15 = lane & 15;
  const int mBase = blockIdx.y * 64 + (wid >> 1) * 16;
  const int nBase = blockIdx.x * 128 + (wid & 1) * 64;
  int ar = mBase + l15;
  if (ar >= M) ar = M - 1;
  const bf16_t* ap = A + (size_t)ar * K;
  const bf16_t* bp0 = Bt + (size_t)(nBase + l15) * K;
  v8f c0 = {}, c1 = {}, c2 = {}, c3 = {};
  for (int kk = 0; kk < K; kk += 32) {
    v16bf a = load_frag(ap + kk, hi);
    c0 = wmma_bf16(a, load_frag(bp0 + kk, hi), c0);
    c1 = wmma_bf16(a, load_frag(bp0 + (size_t)16 * K + kk, hi), c1);
    c2 = wmma_bf16(a, load_frag(bp0 + (size_t)32 * K + kk, hi), c2);
    c3 = wmma_bf16(a, load_frag(bp0 + (size_t)48 * K + kk, hi), c3);
  }
#pragma unroll
  for (int r = 0; r < 8; ++r) {
    int row = mBase + r + hi * 8;
    if (row < M) {
      size_t o = (size_t)row * N + nBase + l15;
      C[o + 0]  = (bf16_t)(c0[r] * scale);
      C[o + 16] = (bf16_t)(c1[r] * scale);
      C[o + 32] = (bf16_t)(c2[r] * scale);
      C[o + 48] = (bf16_t)(c3[r] * scale);
    }
  }
}

// f32 output + bias + residual (in-place residual is safe: 1 writer/element)
__global__ void __launch_bounds__(256) k_gemm_f32_bias_res(
    const bf16_t* __restrict__ A, const bf16_t* __restrict__ Bt,
    float* __restrict__ C, const float* __restrict__ bias,
    const float* __restrict__ res, int M, int N, int K) {
  const int lane = threadIdx.x & 31, wid = threadIdx.x >> 5;
  const int hi = (lane >> 4) & 1, l15 = lane & 15;
  const int mBase = blockIdx.y * 64 + (wid >> 1) * 16;
  const int nBase = blockIdx.x * 128 + (wid & 1) * 64;
  int ar = mBase + l15;
  if (ar >= M) ar = M - 1;
  const bf16_t* ap = A + (size_t)ar * K;
  const bf16_t* bp0 = Bt + (size_t)(nBase + l15) * K;
  v8f c0 = {}, c1 = {}, c2 = {}, c3 = {};
  for (int kk = 0; kk < K; kk += 32) {
    v16bf a = load_frag(ap + kk, hi);
    c0 = wmma_bf16(a, load_frag(bp0 + kk, hi), c0);
    c1 = wmma_bf16(a, load_frag(bp0 + (size_t)16 * K + kk, hi), c1);
    c2 = wmma_bf16(a, load_frag(bp0 + (size_t)32 * K + kk, hi), c2);
    c3 = wmma_bf16(a, load_frag(bp0 + (size_t)48 * K + kk, hi), c3);
  }
  const float b0 = bias[nBase + l15], b1 = bias[nBase + l15 + 16];
  const float b2 = bias[nBase + l15 + 32], b3 = bias[nBase + l15 + 48];
#pragma unroll
  for (int r = 0; r < 8; ++r) {
    int row = mBase + r + hi * 8;
    if (row < M) {
      size_t o = (size_t)row * N + nBase + l15;
      C[o + 0]  = c0[r] + b0 + res[o + 0];
      C[o + 16] = c1[r] + b1 + res[o + 16];
      C[o + 32] = c2[r] + b2 + res[o + 32];
      C[o + 48] = c3[r] + b3 + res[o + 48];
    }
  }
}

// Fused GEGLU: each wave computes matching u / gate tiles (cols c and
// c+Nhalf of A@Wp), applies u*gelu(gate), writes bf16 [M, Nhalf].
__global__ void __launch_bounds__(256) k_gemm_geglu(
    const bf16_t* __restrict__ A, const bf16_t* __restrict__ Bt,
    bf16_t* __restrict__ C, const float* __restrict__ bias,
    int M, int Nhalf, int K) {
  const int lane = threadIdx.x & 31, wid = threadIdx.x >> 5;
  const int hi = (lane >> 4) & 1, l15 = lane & 15;
  const int mBase = blockIdx.y * 64 + (wid >> 1) * 16;
  const int nBase = blockIdx.x * 128 + (wid & 1) * 64;
  int ar = mBase + l15;
  if (ar >= M) ar = M - 1;
  const bf16_t* ap = A + (size_t)ar * K;
  const bf16_t* bu = Bt + (size_t)(nBase + l15) * K;
  const bf16_t* bg = Bt + (size_t)(nBase + Nhalf + l15) * K;
  v8f u0 = {}, u1 = {}, u2 = {}, u3 = {};
  v8f g0 = {}, g1 = {}, g2 = {}, g3 = {};
  for (int kk = 0; kk < K; kk += 32) {
    v16bf a = load_frag(ap + kk, hi);
    u0 = wmma_bf16(a, load_frag(bu + kk, hi), u0);
    u1 = wmma_bf16(a, load_frag(bu + (size_t)16 * K + kk, hi), u1);
    u2 = wmma_bf16(a, load_frag(bu + (size_t)32 * K + kk, hi), u2);
    u3 = wmma_bf16(a, load_frag(bu + (size_t)48 * K + kk, hi), u3);
    g0 = wmma_bf16(a, load_frag(bg + kk, hi), g0);
    g1 = wmma_bf16(a, load_frag(bg + (size_t)16 * K + kk, hi), g1);
    g2 = wmma_bf16(a, load_frag(bg + (size_t)32 * K + kk, hi), g2);
    g3 = wmma_bf16(a, load_frag(bg + (size_t)48 * K + kk, hi), g3);
  }
#pragma unroll
  for (int r = 0; r < 8; ++r) {
    int row = mBase + r + hi * 8;
    if (row < M) {
      size_t o = (size_t)row * Nhalf + nBase + l15;
#pragma unroll
      for (int t = 0; t < 4; ++t) {
        int col = nBase + t * 16 + l15;
        float uv = (t == 0 ? u0[r] : t == 1 ? u1[r] : t == 2 ? u2[r] : u3[r]) + bias[col];
        float gv = (t == 0 ? g0[r] : t == 1 ? g1[r] : t == 2 ? g2[r] : g3[r]) + bias[col + Nhalf];
        float gl = 0.5f * gv * (1.f + erff(gv * 0.70710678118654752f));
        C[o + t * 16] = (bf16_t)(uv * gl);
      }
    }
  }
}

// ---------------------------------------------------------------------------
// Flash attention: one wave per (batch, head, 16-query tile).
// q pre-scaled by 1/sqrt(64). k row-major [b*mRows,1024]; vt [b][h][64][mpad].
// ---------------------------------------------------------------------------
__global__ void __launch_bounds__(256) k_attention(
    const bf16_t* __restrict__ q, const bf16_t* __restrict__ k,
    const bf16_t* __restrict__ vt, bf16_t* __restrict__ out,
    int nq, int m, int mRows, int mpad) {
  __shared__ __align__(16) bf16_t ldsP[8][16][32];
  const int lane = threadIdx.x & 31, wid = threadIdx.x >> 5;
  const int hi = (lane >> 4) & 1, l15 = lane & 15;
  int w = blockIdx.x * 8 + wid;
  if (w >= 2 * 16 * nq) return;
  const int qt = w % nq;
  int t2 = w / nq;
  const int hh = t2 & 15, bb = t2 >> 4;

  const bf16_t* qp = q + (size_t)(bb * 2048 + qt * 16 + l15) * 1024 + hh * 64;
  const v16bf qa0 = load_frag(qp, hi);
  const v16bf qa1 = load_frag(qp + 32, hi);

  v8f o0 = {}, o1 = {}, o2 = {}, o3 = {};
  float mrow[8], lrow[8];
#pragma unroll
  for (int r = 0; r < 8; ++r) { mrow[r] = -1e30f; lrow[r] = 0.f; }

  const bf16_t* vbase = vt + (size_t)(bb * 16 + hh) * 64 * mpad;

  for (int j = 0; j < mpad; j += 32) {
    const bf16_t* kp0 = k + (size_t)(bb * mRows + j + l15) * 1024 + hh * 64;
    const bf16_t* kp1 = kp0 + (size_t)16 * 1024;
    v8f s0 = {}, s1 = {};
    s0 = wmma_bf16(qa0, load_frag(kp0, hi), s0);
    s0 = wmma_bf16(qa1, load_frag(kp0 + 32, hi), s0);
    s1 = wmma_bf16(qa0, load_frag(kp1, hi), s1);
    s1 = wmma_bf16(qa1, load_frag(kp1 + 32, hi), s1);

    const bool v0 = (j + l15) < m, v1 = (j + 16 + l15) < m;
#pragma unroll
    for (int r = 0; r < 8; ++r) {
      float a = v0 ? s0[r] : -1e30f;
      float b = v1 ? s1[r] : -1e30f;
      float rm = fmaxf(a, b);
      rm = fmaxf(rm, __shfl_xor(rm, 1, 32));
      rm = fmaxf(rm, __shfl_xor(rm, 2, 32));
      rm = fmaxf(rm, __shfl_xor(rm, 4, 32));
      rm = fmaxf(rm, __shfl_xor(rm, 8, 32));
      float mnew = fmaxf(mrow[r], rm);
      float alpha = __expf(mrow[r] - mnew);
      float p0 = __expf(a - mnew);
      float p1 = __expf(b - mnew);
      float rs = p0 + p1;
      rs += __shfl_xor(rs, 1, 32);
      rs += __shfl_xor(rs, 2, 32);
      rs += __shfl_xor(rs, 4, 32);
      rs += __shfl_xor(rs, 8, 32);
      lrow[r] = lrow[r] * alpha + rs;
      mrow[r] = mnew;
      o0[r] *= alpha; o1[r] *= alpha; o2[r] *= alpha; o3[r] *= alpha;
      const int prow = r + hi * 8;
      ldsP[wid][prow][l15] = (bf16_t)p0;
      ldsP[wid][prow][l15 + 16] = (bf16_t)p1;
    }
    // Re-stripe P (C-layout) into an A-fragment via LDS (same-wave DS ops
    // are in-order; compiler inserts s_wait_dscnt for the RAW).
    const v16bf pa = load_frag(&ldsP[wid][l15][0], hi);
    const bf16_t* vp = vbase + (size_t)l15 * mpad + j;
    o0 = wmma_bf16(pa, load_frag(vp, hi), o0);
    o1 = wmma_bf16(pa, load_frag(vp + (size_t)16 * mpad, hi), o1);
    o2 = wmma_bf16(pa, load_frag(vp + (size_t)32 * mpad, hi), o2);
    o3 = wmma_bf16(pa, load_frag(vp + (size_t)48 * mpad, hi), o3);
  }

  bf16_t* op = out + (size_t)(bb * 2048 + qt * 16) * 1024 + hh * 64;
#pragma unroll
  for (int r = 0; r < 8; ++r) {
    const int row = r + hi * 8;
    const float inv = 1.0f / lrow[r];
    op[(size_t)row * 1024 + l15 + 0]  = (bf16_t)(o0[r] * inv);
    op[(size_t)row * 1024 + l15 + 16] = (bf16_t)(o1[r] * inv);
    op[(size_t)row * 1024 + l15 + 32] = (bf16_t)(o2[r] * inv);
    op[(size_t)row * 1024 + l15 + 48] = (bf16_t)(o3[r] * inv);
  }
}

// ---------------------------------------------------------------------------
extern "C" void kernel_launch(void* const* d_in, const int* in_sizes, int n_in,
                              void* d_out, int out_size, void* d_ws, size_t ws_size,
                              hipStream_t stream) {
  (void)in_sizes; (void)n_in; (void)out_size; (void)ws_size;
  const float* x    = (const float*)d_in[0];
  const float* ctx  = (const float*)d_in[1];
  const float* ln1g = (const float*)d_in[2];
  const float* ln1b = (const float*)d_in[3];
  const float* a1Wq = (const float*)d_in[4];
  const float* a1Wk = (const float*)d_in[5];
  const float* a1Wv = (const float*)d_in[6];
  const float* a1Wo = (const float*)d_in[7];
  const float* a1bo = (const float*)d_in[8];
  const float* ln2g = (const float*)d_in[9];
  const float* ln2b = (const float*)d_in[10];
  const float* a2Wq = (const float*)d_in[11];
  const float* a2Wk = (const float*)d_in[12];
  const float* a2Wv = (const float*)d_in[13];
  const float* a2Wo = (const float*)d_in[14];
  const float* a2bo = (const float*)d_in[15];
  const float* ln3g = (const float*)d_in[16];
  const float* ln3b = (const float*)d_in[17];
  const float* ffWp = (const float*)d_in[18];
  const float* ffbp = (const float*)d_in[19];
  const float* ffWo = (const float*)d_in[20];
  const float* ffbo = (const float*)d_in[21];
  float* out = (float*)d_out;

  char* ws = (char*)d_ws;
  size_t off = 0;
  auto al = [&](size_t bytes) {
    size_t r = off;
    off = (off + bytes + 255) & ~(size_t)255;
    return r;
  };
  const size_t MM = 1024 * 1024;
  const size_t oWq1 = al(MM * 2), oWk1 = al(MM * 2), oWv1 = al(MM * 2), oWo1 = al(MM * 2);
  const size_t oWq2 = al(MM * 2), oWk2 = al(768 * 1024 * 2), oWv2 = al(768 * 1024 * 2), oWo2 = al(MM * 2);
  const size_t oWp = al((size_t)8192 * 1024 * 2), oWoF = al((size_t)4096 * 1024 * 2);
  const size_t oX = al((size_t)4096 * 1024 * 4);
  const size_t oH = al((size_t)4096 * 1024 * 2);
  const size_t oQ = al((size_t)4096 * 1024 * 2);
  const size_t oK = al((size_t)4096 * 1024 * 2);
  const size_t oV = al((size_t)4096 * 1024 * 2);
  const size_t oVt = al((size_t)2 * 16 * 64 * 2048 * 2);
  const size_t oAt = al((size_t)4096 * 1024 * 2);
  const size_t oCtx = al((size_t)154 * 768 * 2);

  bf16_t* Wq1 = (bf16_t*)(ws + oWq1); bf16_t* Wk1 = (bf16_t*)(ws + oWk1);
  bf16_t* Wv1 = (bf16_t*)(ws + oWv1); bf16_t* Wo1 = (bf16_t*)(ws + oWo1);
  bf16_t* Wq2 = (bf16_t*)(ws + oWq2); bf16_t* Wk2 = (bf16_t*)(ws + oWk2);
  bf16_t* Wv2 = (bf16_t*)(ws + oWv2); bf16_t* Wo2 = (bf16_t*)(ws + oWo2);
  bf16_t* WpT = (bf16_t*)(ws + oWp);  bf16_t* WoF = (bf16_t*)(ws + oWoF);
  float*  xcur = (float*)(ws + oX);
  bf16_t* hbuf = (bf16_t*)(ws + oH);
  bf16_t* qb = (bf16_t*)(ws + oQ);
  bf16_t* kb = (bf16_t*)(ws + oK);
  bf16_t* vb = (bf16_t*)(ws + oV);
  bf16_t* vtb = (bf16_t*)(ws + oVt);
  bf16_t* atb = (bf16_t*)(ws + oAt);
  bf16_t* ctxb = (bf16_t*)(ws + oCtx);
  bf16_t* ffb = qb;  // alias: FF activations reuse q/k/v/vt (32 MiB), phases ordered

  const dim3 blk(256);
  const float SCALE = 0.125f;  // 64^-0.5, exact in bf16

  // 1) weights -> bf16 transposed
  k_wt<<<4096, blk, 0, stream>>>(a1Wq, Wq1, 1024, 1024);
  k_wt<<<4096, blk, 0, stream>>>(a1Wk, Wk1, 1024, 1024);
  k_wt<<<4096, blk, 0, stream>>>(a1Wv, Wv1, 1024, 1024);
  k_wt<<<4096, blk, 0, stream>>>(a1Wo, Wo1, 1024, 1024);
  k_wt<<<4096, blk, 0, stream>>>(a2Wq, Wq2, 1024, 1024);
  k_wt<<<3072, blk, 0, stream>>>(a2Wk, Wk2, 768, 1024);
  k_wt<<<3072, blk, 0, stream>>>(a2Wv, Wv2, 768, 1024);
  k_wt<<<4096, blk, 0, stream>>>(a2Wo, Wo2, 1024, 1024);
  k_wt<<<32768, blk, 0, stream>>>(ffWp, WpT, 1024, 8192);
  k_wt<<<16384, blk, 0, stream>>>(ffWo, WoF, 4096, 1024);

  const dim3 gP(8, 64);  // M=4096, N=1024 GEMMs

  // 2) self-attention
  k_layernorm<<<4096, blk, 0, stream>>>(x, ln1g, ln1b, hbuf);
  k_gemm_bf16<<<gP, blk, 0, stream>>>(hbuf, Wq1, qb, 4096, 1024, 1024, SCALE);
  k_gemm_bf16<<<gP, blk, 0, stream>>>(hbuf, Wk1, kb, 4096, 1024, 1024, 1.0f);
  k_gemm_bf16<<<gP, blk, 0, stream>>>(hbuf, Wv1, vb, 4096, 1024, 1024, 1.0f);
  k_vt<<<(2 * 16 * 64 * 2048) / 256, blk, 0, stream>>>(vb, vtb, 2048, 2048);
  k_attention<<<512, blk, 0, stream>>>(qb, kb, vtb, atb, 128, 2048, 2048, 2048);
  k_gemm_f32_bias_res<<<gP, blk, 0, stream>>>(atb, Wo1, xcur, a1bo, x, 4096, 1024, 1024);

  // 3) cross-attention
  k_layernorm<<<4096, blk, 0, stream>>>(xcur, ln2g, ln2b, hbuf);
  k_gemm_bf16<<<gP, blk, 0, stream>>>(hbuf, Wq2, qb, 4096, 1024, 1024, SCALE);
  k_cvt<<<(154 * 768 + 255) / 256, blk, 0, stream>>>(ctx, ctxb, 154 * 768);
  {
    dim3 gKV(8, 3);  // M=154
    k_gemm_bf16<<<gKV, blk, 0, stream>>>(ctxb, Wk2, kb, 154, 1024, 768, 1.0f);
    k_gemm_bf16<<<gKV, blk, 0, stream>>>(ctxb, Wv2, vb, 154, 1024, 768, 1.0f);
  }
  k_vt<<<(2 * 16 * 64 * 96 + 255) / 256, blk, 0, stream>>>(vb, vtb, 77, 96);
  k_attention<<<512, blk, 0, stream>>>(qb, kb, vtb, atb, 128, 77, 77, 96);
  k_gemm_f32_bias_res<<<gP, blk, 0, stream>>>(atb, Wo2, xcur, a2bo, xcur, 4096, 1024, 1024);

  // 4) GEGLU feed-forward
  k_layernorm<<<4096, blk, 0, stream>>>(xcur, ln3g, ln3b, hbuf);
  {
    dim3 gFF(32, 64);  // Nhalf=4096
    k_gemm_geglu<<<gFF, blk, 0, stream>>>(hbuf, WpT, ffb, ffbp, 4096, 4096, 1024);
  }
  k_gemm_f32_bias_res<<<gP, blk, 0, stream>>>(ffb, WoF, out, ffbo, xcur, 4096, 1024, 4096);
}
